// CifarBaselineSNN_35914516529253
// MI455X (gfx1250) — matrix operand order repaired and checked by reading, not compiled, not run on hardware
//
#include <hip/hip_runtime.h>

typedef __attribute__((ext_vector_type(16))) _Float16 v16h;
typedef __attribute__((ext_vector_type(8)))  _Float16 h8;
typedef __attribute__((ext_vector_type(8)))  float    v8f;
typedef int v4i_gcc __attribute__((vector_size(16)));   // async-LDS builtin param type

#define T_STEPS 8
#define BATCH   128

#if __has_builtin(__builtin_amdgcn_global_load_async_to_lds_b128)
#define HAVE_ASYNC_LDS 1
#else
#define HAVE_ASYNC_LDS 0
#endif

__device__ __forceinline__ void wait_async0() {
#if __has_builtin(__builtin_amdgcn_s_wait_asynccnt)
  __builtin_amdgcn_s_wait_asynccnt(0);
#else
  asm volatile("s_wait_asynccnt 0x0" ::: "memory");
#endif
}

// =========================================================================
// Prep kernels: pad/convert activations+weights to f16 once per launch.
// =========================================================================

// x [T*B,3,32,32] f32 NCHW -> xp [T*B, 34*34, 4] f16 NHWC, zero halo and a
// zero 4th-channel plane (absorbs conv1's K padding 27->32 branch-free).
__global__ __launch_bounds__(256) void prep_pad_x(const float* __restrict__ x,
                                                  _Float16* __restrict__ xp) {
  const int idx = blockIdx.x * 256 + threadIdx.x;   // 1024*1156*4
  const int c = idx & 3;
  const int q = idx >> 2;
  const int pos = q % 1156;
  const int tb  = q / 1156;
  const int hh = pos / 34, ww = pos - hh * 34;
  float v = 0.f;
  if (c < 3 && hh >= 1 && hh <= 32 && ww >= 1 && ww <= 32)
    v = x[(((long)tb * 3 + c) << 10) + ((hh - 1) << 5) + (ww - 1)];
  xp[idx] = (_Float16)v;
}

// w1 [64,27] f32 -> w1p [64,32] f16, K zero-padded.
__global__ __launch_bounds__(256) void prep_w1(const float* __restrict__ w1,
                                               _Float16* __restrict__ w1p) {
  const int idx = blockIdx.x * 256 + threadIdx.x;   // 2048
  const int oc = idx >> 5, k = idx & 31;
  w1p[idx] = (_Float16)((k < 27) ? w1[oc * 27 + k] : 0.f);
}

// w2 [128,576] f32 -> w2s [18][8][32][16] f16: exact per-lane WMMA-B
// fragment image per 32-K chunk (lane l: col=l&15, K=(l>>4)*16+j).
__global__ __launch_bounds__(256) void prep_w2s(const float* __restrict__ w2,
                                                _Float16* __restrict__ w2s) {
  const int idx = blockIdx.x * 256 + threadIdx.x;   // 73728
  const int j  = idx & 15;
  int q = idx >> 4;
  const int ln = q & 31;  q >>= 5;
  const int nt = q & 7;
  const int kc = q >> 3;
  const int k  = kc * 32 + ((ln >> 4) << 4) + j;
  const int oc = nt * 16 + (ln & 15);
  w2s[idx] = (_Float16)w2[oc * 576 + k];
}

// fc1_w [256,8192] f32 (K indexed c*64+pix) -> f16 with K permuted to the
// NHWC feature order (pix*128+c) used by the stage-2 spike tensor.
__global__ __launch_bounds__(256) void prep_fc1w(const float* __restrict__ w,
                                                 _Float16* __restrict__ w16) {
  const int idx = blockIdx.x * 256 + threadIdx.x;   // 2097152
  const int o = idx >> 13;
  const int r = idx & 8191;          // output k' = pix*128 + c
  const int pix = r >> 7, c = r & 127;
  w16[idx] = (_Float16)w[o * 8192 + c * 64 + pix];
}

__global__ void zero_h8(h8* __restrict__ p, long n) {
  long i = (long)blockIdx.x * 256 + threadIdx.x;
  h8 z = {};
  if (i < n) p[i] = z;
}

__global__ void zero_f32(float* __restrict__ p, int n) {
  int i = blockIdx.x * 256 + threadIdx.x;
  if (i < n) p[i] = 0.f;
}

// =========================================================================
// conv1: implicit GEMM, M = T*B*32*32, N = 64, K = 32 (padded). One wave =
// 16 M rows; A gather from padded NHWC xp (lane stride 8B -> coalesced);
// B fragments are contiguous 32B vector loads from w1p.
// =========================================================================
__global__ __launch_bounds__(256) void conv1_wmma(const _Float16* __restrict__ xp,
                                                  const _Float16* __restrict__ w1p,
                                                  _Float16* __restrict__ y1) {
  const int lane = threadIdx.x & 31;
  const int wave = threadIdx.x >> 5;
  const int lg = lane >> 4, lr = lane & 15;
  const long rowBase = (long)blockIdx.x * 128 + wave * 16;
  const int p = (int)rowBase + lr;
  const int wpix = p & 31, hpix = (p >> 5) & 31, tb = p >> 10;

  const _Float16* abase = xp + ((long)tb * 1156 + hpix * 34 + wpix) * 4;
  v16h a;
#pragma unroll
  for (int j = 0; j < 16; ++j) {
    int k = (lg << 4) + j;   // 0..31
    int ci = k / 9, r = k - ci * 9, kh = r / 3, kw = r - kh * 3;
    a[j] = abase[(kh * 34 + kw) * 4 + ci];           // k>=27 hits zero plane
  }

#pragma unroll
  for (int nt = 0; nt < 4; ++nt) {
    const int oc = nt * 16 + lr;
    v16h bm = *(const v16h*)(w1p + oc * 32 + (lg << 4));
    v8f acc = {};
    acc = __builtin_amdgcn_wmma_f32_16x16x32_f16(false, a, false, bm,
                                                 (short)0, acc, false, false);
#pragma unroll
    for (int r = 0; r < 8; ++r) {
      long m = rowBase + (lg << 3) + r;
      int ww2 = (int)(m & 31), hh2 = (int)((m >> 5) & 31);
      long tb2 = m >> 10;
      // NHWC store: consecutive lanes -> consecutive channels (coalesced)
      y1[((tb2 << 10) + (hh2 << 5) + ww2) * 64 + oc] = (_Float16)acc[r];
    }
  }
}

// =========================================================================
// Per-channel sum/sumsq over NHWC [rows][C]; channel-fastest threads keep
// every global read fully coalesced. Block partial -> 2 atomics/channel.
// =========================================================================
__global__ __launch_bounds__(256) void channel_stats(const _Float16* __restrict__ y,
                                                     float* __restrict__ stats,
                                                     int C, int rowsPerBlock,
                                                     long totalRows) {
  const int G = 256 / C;                 // row-groups per pass (4 or 2)
  const int c = threadIdx.x % C;
  const int g = threadIdx.x / C;
  long r = (long)blockIdx.x * rowsPerBlock + g;
  long rEnd = (long)(blockIdx.x + 1) * rowsPerBlock;
  if (rEnd > totalRows) rEnd = totalRows;
  float s = 0.f, ss = 0.f;
  for (; r < rEnd; r += G) {
    float v = (float)y[r * C + c];
    s += v; ss += v * v;
  }
  __shared__ float red0[256];
  __shared__ float red1[256];
  red0[threadIdx.x] = s; red1[threadIdx.x] = ss;
  __syncthreads();
  if (g == 0) {
    for (int i = 1; i < G; ++i) { s += red0[c + i * C]; ss += red1[c + i * C]; }
    atomicAdd(&stats[c],     s);
    atomicAdd(&stats[C + c], ss);
  }
}

// =========================================================================
// Fused BN + LIF (T loop, 4 membrane states in regs) + 2x2 avg-pool.
// NHWC input (channel-fastest threads -> coalesced reads). Output either
// NHWC (stage 2) or zero-haloed NCHW (stage 1, feeds conv2's gather).
// =========================================================================
__global__ __launch_bounds__(256) void bn_lif_pool(const _Float16* __restrict__ y,
                                                   _Float16* __restrict__ s_out,
                                                   const float* __restrict__ stats,
                                                   const float* __restrict__ gamma,
                                                   const float* __restrict__ beta,
                                                   int C, int W, int planeIn,
                                                   int planeOut, int rowOut,
                                                   int outOff, int outNCHW,
                                                   float invCount) {
  const int Wp = W >> 1, Hp = W >> 1;
  int idx = blockIdx.x * 256 + threadIdx.x;
  const int c = idx % C;  idx /= C;
  const int wo = idx % Wp; idx /= Wp;
  const int ho = idx % Hp;
  const int b  = idx / Hp;

  const float mean = stats[c] * invCount;
  const float var  = stats[C + c] * invCount - mean * mean;
  const float g  = gamma[c] * rsqrtf(var + 1e-5f);
  const float sh = beta[c] - mean * g;

  const int pos = outOff + ho * rowOut + wo;
  float v0 = 0.f, v1 = 0.f, v2 = 0.f, v3 = 0.f;
  for (int t = 0; t < T_STEPS; ++t) {
    const long tb = t * BATCH + b;
    const _Float16* base = y + (tb * planeIn + (2 * ho) * W + 2 * wo) * C + c;
    float x0 = (float)base[0]           * g + sh;
    float x1 = (float)base[C]           * g + sh;
    float x2 = (float)base[W * C]       * g + sh;
    float x3 = (float)base[(W + 1) * C] * g + sh;
    v0 += (x0 - v0) * 0.5f;  v1 += (x1 - v1) * 0.5f;
    v2 += (x2 - v2) * 0.5f;  v3 += (x3 - v3) * 0.5f;
    float s0 = (v0 >= 1.f) ? 1.f : 0.f;
    float s1 = (v1 >= 1.f) ? 1.f : 0.f;
    float s2 = (v2 >= 1.f) ? 1.f : 0.f;
    float s3 = (v3 >= 1.f) ? 1.f : 0.f;
    v0 *= (1.f - s0); v1 *= (1.f - s1); v2 *= (1.f - s2); v3 *= (1.f - s3);
    long o = outNCHW ? ((tb * C + c) * planeOut + pos)
                     : ((tb * planeOut + pos) * C + c);
    s_out[o] = (_Float16)((s0 + s1 + s2 + s3) * 0.25f);
  }
}

// =========================================================================
// conv2: implicit GEMM, M = 262144, N = 128, K = 576 = 18*32.
// 8 waves/block (128 M rows); each wave holds all 8 N-tiles in accumulators
// so one A gather feeds 8 WMMAs. A gather from zero-haloed NCHW spike map
// (lane stride 2B -> coalesced). B chunks stream through a double-buffered
// LDS tile via async load-to-LDS, overlapped with compute.
// =========================================================================
__global__ __launch_bounds__(256) void conv2_wmma(const _Float16* __restrict__ s1p,
                                                  const _Float16* __restrict__ w2s,
                                                  _Float16* __restrict__ y2) {
  __shared__ __align__(32) _Float16 Bs[2][8][32][16];  // [buf][ntile][lane][half]

  const int tid  = threadIdx.x;
  const int lane = tid & 31, wave = tid >> 5;
  const int lg = lane >> 4, lr = lane & 15;
  const long rowBase = (long)blockIdx.x * 128 + wave * 16;
  const int p = (int)rowBase + lr;
  const int wpix = p & 15, hpix = (p >> 4) & 15, tb = p >> 8;

  // NCHW padded spike map: [tb][64][18][18]
  const _Float16* abase = s1p + (long)tb * 20736 + hpix * 18 + wpix;
  v8f acc[8] = {};

  auto stage = [&](int kc, int buf) {
    const _Float16* gsrc = w2s + kc * 4096 + tid * 16;
    _Float16* dst = &Bs[buf][0][0][0] + tid * 16;
#if HAVE_ASYNC_LDS
    __builtin_amdgcn_global_load_async_to_lds_b128(
        (v4i_gcc*)gsrc,       (v4i_gcc*)dst,       0, 0);
    __builtin_amdgcn_global_load_async_to_lds_b128(
        (v4i_gcc*)(gsrc + 8), (v4i_gcc*)(dst + 8), 0, 0);
#else
    *(h8*)dst       = *(const h8*)gsrc;
    *(h8*)(dst + 8) = *(const h8*)(gsrc + 8);
#endif
  };

  stage(0, 0);                          // prologue: chunk 0 -> buffer 0
  for (int kc = 0; kc < 18; ++kc) {
    const int cur = kc & 1;
#if HAVE_ASYNC_LDS
    wait_async0();                      // my async writes into Bs[cur] done
#endif
    __syncthreads();                    // everyone's writes visible; prior reads done
    if (kc < 17) stage(kc + 1, 1 - cur);

    // Branch-free, lane-coalesced A gather
    v16h a;
#pragma unroll
    for (int j = 0; j < 16; ++j) {
      int k = kc * 32 + (lg << 4) + j;
      int ci = k / 9, r = k - ci * 9, kh = r / 3, kw = r - kh * 3;
      a[j] = abase[ci * 324 + kh * 18 + kw];
    }

#pragma unroll
    for (int nt = 0; nt < 8; ++nt) {
      v16h bm = *(const v16h*)&Bs[cur][nt][lane][0];
      acc[nt] = __builtin_amdgcn_wmma_f32_16x16x32_f16(false, a, false, bm,
                                                       (short)0, acc[nt],
                                                       false, false);
    }
  }

#pragma unroll
  for (int nt = 0; nt < 8; ++nt) {
    const int oc = nt * 16 + lr;
#pragma unroll
    for (int r = 0; r < 8; ++r) {
      long m = rowBase + (lg << 3) + r;
      int ww2 = (int)(m & 15), hh2 = (int)((m >> 4) & 15);
      long tb2 = m >> 8;
      y2[((tb2 << 8) + (hh2 << 4) + ww2) * 128 + oc] = (_Float16)acc[nt][r];
    }
  }
}

// =========================================================================
// fc1: M = 1024, N = 256, K = 8192. A rows contiguous in K (two 16B loads
// per lane); B fragments are single 32B vector loads from pre-permuted f16.
// =========================================================================
__global__ __launch_bounds__(256) void fc1_wmma(const _Float16* __restrict__ a_mat,
                                                const _Float16* __restrict__ w16,
                                                float* __restrict__ out) {
  const int tid  = threadIdx.x;
  const int lane = tid & 31, wave = tid >> 5;
  const int lg = lane >> 4, lr = lane & 15;
  const int tileId = blockIdx.x * 8 + wave;      // 64 mtiles x 16 ntiles
  const int mtile = tileId >> 4, ntile = tileId & 15;
  const _Float16* arow = a_mat + (long)(mtile * 16 + lr) * 8192;
  const _Float16* wrow = w16   + (long)(ntile * 16 + lr) * 8192;

  v8f acc = {};
  for (int kc = 0; kc < 8192; kc += 32) {
    h8 a0 = *(const h8*)(arow + kc + (lg << 4));
    h8 a1 = *(const h8*)(arow + kc + 8 + (lg << 4));
    v16h a;
#pragma unroll
    for (int j = 0; j < 8; ++j) { a[j] = a0[j]; a[j + 8] = a1[j]; }
    v16h bm = *(const v16h*)(wrow + kc + (lg << 4));
    acc = __builtin_amdgcn_wmma_f32_16x16x32_f16(false, a, false, bm,
                                                 (short)0, acc, false, false);
  }
#pragma unroll
  for (int r = 0; r < 8; ++r) {
    int m = mtile * 16 + (lg << 3) + r;
    out[(long)m * 256 + ntile * 16 + lr] = acc[r];
  }
}

// LIF over T for fc1 pre-activations: one thread per (b, o).
__global__ __launch_bounds__(256) void lif_fc(const float* __restrict__ pre,
                                              _Float16* __restrict__ sp) {
  const int idx = blockIdx.x * 256 + threadIdx.x;   // 32768
  const int b = idx >> 8, o = idx & 255;
  float v = 0.f;
  for (int t = 0; t < T_STEPS; ++t) {
    long off = ((long)(t * BATCH + b) << 8) + o;
    float x = pre[off];
    v += (x - v) * 0.5f;
    float s = (v >= 1.f) ? 1.f : 0.f;
    sp[off] = (_Float16)s;
    v *= (1.f - s);
  }
}

// fc2 (256 -> 10) + bias.
__global__ __launch_bounds__(256) void fc2_out(const _Float16* __restrict__ sp,
                                               const float* __restrict__ w,
                                               const float* __restrict__ bias,
                                               float* __restrict__ out) {
  const int idx = blockIdx.x * 256 + threadIdx.x;
  if (idx >= T_STEPS * BATCH * 10) return;
  const int tb = idx / 10, o = idx - tb * 10;
  const _Float16* a = sp + (long)tb * 256;
  const float*   wr = w + o * 256;
  float s = bias[o];
  for (int i = 0; i < 256; ++i) s += (float)a[i] * wr[i];
  out[idx] = s;
}

// =========================================================================
extern "C" void kernel_launch(void* const* d_in, const int* in_sizes, int n_in,
                              void* d_out, int out_size, void* d_ws, size_t ws_size,
                              hipStream_t stream) {
  (void)in_sizes; (void)n_in; (void)out_size; (void)ws_size;
  const float* x     = (const float*)d_in[0];
  const float* w1    = (const float*)d_in[1];
  const float* bn1g  = (const float*)d_in[2];
  const float* bn1b  = (const float*)d_in[3];
  const float* w2    = (const float*)d_in[4];
  const float* bn2g  = (const float*)d_in[5];
  const float* bn2b  = (const float*)d_in[6];
  const float* fc1w  = (const float*)d_in[7];
  const float* fc2w  = (const float*)d_in[8];
  const float* fc2b  = (const float*)d_in[9];

  // workspace carve-up (all region sizes multiples of 16 halves -> 32B align)
  _Float16* xp   = (_Float16*)d_ws;        // [1024][1156][4]   4,734,976
  _Float16* w1p  = xp   + 4734976;         // [64][32]              2,048
  _Float16* y1   = w1p  + 2048;            // [1024][1024][64] 67,108,864
  _Float16* s1p  = y1   + 67108864;        // [1024][64][324]  21,233,664 (NCHW padded)
  _Float16* y2   = s1p  + 21233664;        // [1024][256][128] 33,554,432
  _Float16* s2   = y2   + 33554432;        // [1024][64][128]   8,388,608
  _Float16* w2s  = s2   + 8388608;         // [18][8][32][16]      73,728
  _Float16* fw16 = w2s  + 73728;           // [256][8192]       2,097,152
  _Float16* s3   = fw16 + 2097152;         // [1024][256]         262,144
  float*    f1o  = (float*)(s3 + 262144);  // [1024][256]         262,144
  float*    st1  = f1o + 262144;           // 2*64
  float*    st2  = st1 + 128;              // 2*128

  // prep (once per launch, deterministic)
  prep_pad_x<<<18496, 256, 0, stream>>>(x, xp);
  prep_w1<<<8, 256, 0, stream>>>(w1, w1p);
  prep_w2s<<<288, 256, 0, stream>>>(w2, w2s);
  prep_fc1w<<<8192, 256, 0, stream>>>(fc1w, fw16);
  zero_f32<<<2, 256, 0, stream>>>(st1, 384);
  zero_h8<<<10368, 256, 0, stream>>>((h8*)s1p, 21233664 / 8);  // halo zeros

  // layer 1
  conv1_wmma<<<8192, 256, 0, stream>>>(xp, w1p, y1);
  channel_stats<<<2048, 256, 0, stream>>>(y1, st1, 64, 512, 1048576);
  bn_lif_pool<<<8192, 256, 0, stream>>>(y1, s1p, st1, bn1g, bn1b,
                                        64, 32, 1024, 324, 18, 19, 1,
                                        1.f / 1048576.f);
  // layer 2
  conv2_wmma<<<2048, 256, 0, stream>>>(s1p, w2s, y2);
  channel_stats<<<512, 256, 0, stream>>>(y2, st2, 128, 512, 262144);
  bn_lif_pool<<<4096, 256, 0, stream>>>(y2, s2, st2, bn2g, bn2b,
                                        128, 16, 256, 64, 8, 0, 0,
                                        1.f / 262144.f);
  // classifier
  fc1_wmma<<<128, 256, 0, stream>>>(s2, fw16, f1o);
  lif_fc<<<128, 256, 0, stream>>>(f1o, s3);
  fc2_out<<<40, 256, 0, stream>>>(s3, fc2w, fc2b, (float*)d_out);
}
